// CaiQuantLinear_28458453303393
// MI455X (gfx1250) — matrix operand order repaired and checked by reading, not compiled, not run on hardware
//
#include <hip/hip_runtime.h>
#include <hip/hip_bf16.h>
#include <stdint.h>

typedef __attribute__((ext_vector_type(16))) _Float16 v16h;
typedef __attribute__((ext_vector_type(8)))  float    v8f;

#define TOKENS 8192
#define KDIM   4096
#define NDIM   4096
#define BM 128
#define BN 128
#define BK 128          // == groupsize: scale/zero constant per K-tile
#define LDT 136         // padded LDS stride in halfs (272 B -> bank rotation)
#define NT  (KDIM / BK) // 32 K-tiles

union FragU { uint4 q[2]; v16h h; };

// A fragment 16x32 f16: lane m (0..15) rows; lanes 16..31 same rows, K+8 bias.
// V0..V3 = K[kh*8 .. kh*8+7], V4..V7 = K[16+kh*8 .. 16+kh*8+7]  (kh = lane>>4)
__device__ __forceinline__ v16h load_frag_a(const _Float16* lds, int mbase, int kb, int lane) {
    const int row = mbase + (lane & 15);
    const int kh  = (lane >> 4) << 3;               // 0 or 8
    const _Float16* base = lds + row * LDT + kb + kh;
    FragU f;
    f.q[0] = *reinterpret_cast<const uint4*>(base);
    f.q[1] = *reinterpret_cast<const uint4*>(base + 16);
    return f.h;
}

// B fragment 32x16 f16: lane n (0..15) cols; V0..V3 = K[kh*8 .. kh*8+7],
// V4..V7 = K[16+kh*8 .. 16+kh*8+7] (kh = lane>>4) -> column-major LDS tile.
__device__ __forceinline__ v16h load_frag_b(const _Float16* lds, int nbase, int kb, int lane) {
    const int col = nbase + (lane & 15);
    const int k   = kb + ((lane >> 4) << 3);        // 0 or 8
    const _Float16* base = lds + col * LDT + k;
    FragU f;
    f.q[0] = *reinterpret_cast<const uint4*>(base);
    f.q[1] = *reinterpret_cast<const uint4*>(base + 16);
    return f.h;
}

// ---- staged global loads (software pipeline) ----

template<bool XHALF>
__device__ __forceinline__ void load_stage_a(uint4 (&st)[8], const void* xin,
                                             size_t rowoff, int k0) {
    if constexpr (XHALF) {
        const uint4* p = reinterpret_cast<const uint4*>(
            (const _Float16*)xin + rowoff + k0);
        #pragma unroll
        for (int i = 0; i < 8; ++i) st[i] = p[i];
    } else {
        const float* p = (const float*)xin + rowoff + k0;
        #pragma unroll
        for (int i = 0; i < 8; ++i) {
            float4 f0 = *reinterpret_cast<const float4*>(p + i * 8);
            float4 f1 = *reinterpret_cast<const float4*>(p + i * 8 + 4);
            union { _Float16 h[8]; uint4 u; } c;
            c.h[0] = (_Float16)f0.x; c.h[1] = (_Float16)f0.y;
            c.h[2] = (_Float16)f0.z; c.h[3] = (_Float16)f0.w;
            c.h[4] = (_Float16)f1.x; c.h[5] = (_Float16)f1.y;
            c.h[6] = (_Float16)f1.z; c.h[7] = (_Float16)f1.w;
            st[i] = c.u;
        }
    }
}

__device__ __forceinline__ void store_stage_a(_Float16* buf, int arow, int akh,
                                              const uint4 (&st)[8]) {
    uint4* d = reinterpret_cast<uint4*>(buf + arow * LDT + akh);
    #pragma unroll
    for (int i = 0; i < 8; ++i) d[i] = st[i];
}

__device__ __forceinline__ void load_stage_b(uint64_t (&w)[4], float& s, int& z,
                                             const uint64_t* __restrict__ qweight,
                                             const float* __restrict__ scales,
                                             const uint64_t* __restrict__ qzeros,
                                             int tile, int bw0, int ng) {
    const size_t base = (size_t)(tile * 8 + bw0) * NDIM + ng;
    #pragma unroll
    for (int i = 0; i < 4; ++i) w[i] = qweight[base + (size_t)i * NDIM];
    s = scales[(size_t)tile * NDIM + ng];
    z = (int)((qzeros[(size_t)tile * (NDIM / 16) + (ng >> 4)]
               >> ((ng & 15) * 4)) & 15) + 1;
}

__device__ __forceinline__ void store_stage_b(_Float16* buf, int bcol, int bw0,
                                              const uint64_t (&w)[4], float s, int z) {
    const float sz = s * (float)z;                  // w = s*q - s*z
    #pragma unroll
    for (int i = 0; i < 4; ++i) {
        union { _Float16 h[16]; uint4 u[2]; } c;
        #pragma unroll
        for (int j = 0; j < 16; ++j) {
            const int q = (int)((w[i] >> (4 * j)) & 15);
            c.h[j] = (_Float16)(s * (float)q - sz);
        }
        uint4* d = reinterpret_cast<uint4*>(buf + bcol * LDT + ((bw0 + i) << 4));
        d[0] = c.u[0];
        d[1] = c.u[1];
    }
}

__device__ __forceinline__ void compute_tile(const _Float16* bufA, const _Float16* bufB,
                                             int wm, int wn, int lane, v8f (&acc)[4][2]) {
    #pragma unroll
    for (int kk = 0; kk < BK; kk += 32) {
        v16h bf0 = load_frag_b(bufB, wn + 0,  kk, lane);
        v16h bf1 = load_frag_b(bufB, wn + 16, kk, lane);
        #pragma unroll
        for (int mi = 0; mi < 4; ++mi) {
            v16h af = load_frag_a(bufA, wm + (mi << 4), kk, lane);
            acc[mi][0] = __builtin_amdgcn_wmma_f32_16x16x32_f16(
                false, af, false, bf0, (short)0, acc[mi][0], false, false);
            acc[mi][1] = __builtin_amdgcn_wmma_f32_16x16x32_f16(
                false, af, false, bf1, (short)0, acc[mi][1], false, false);
        }
    }
}

// ---- pre-pass: x fp32 -> f16 into workspace ----
__global__ __launch_bounds__(256)
void x_to_f16_kernel(const float* __restrict__ x, _Float16* __restrict__ xh) {
    const int idx = blockIdx.x * 256 + threadIdx.x;   // 8 floats per thread
    const float4* src = reinterpret_cast<const float4*>(x) + (size_t)idx * 2;
    float4 f0 = src[0], f1 = src[1];
    union { _Float16 h[8]; uint4 u; } c;
    c.h[0] = (_Float16)f0.x; c.h[1] = (_Float16)f0.y;
    c.h[2] = (_Float16)f0.z; c.h[3] = (_Float16)f0.w;
    c.h[4] = (_Float16)f1.x; c.h[5] = (_Float16)f1.y;
    c.h[6] = (_Float16)f1.z; c.h[7] = (_Float16)f1.w;
    reinterpret_cast<uint4*>(xh)[idx] = c.u;
}

// ---- main GEMM: double-buffered LDS, software-pipelined global loads ----
template<bool XHALF>
__global__ __launch_bounds__(256)
void gptq_wmma_kernel(const void*     __restrict__ xin,      // f16 (ws) or f32
                      const uint64_t* __restrict__ qweight,  // [KDIM/16, NDIM]
                      const float*    __restrict__ scales,   // [NT, NDIM]
                      const uint64_t* __restrict__ qzeros,   // [NT, NDIM/16]
                      const float*    __restrict__ bias,     // [NDIM]
                      float*          __restrict__ out)      // [TOKENS, NDIM]
{
    __shared__ _Float16 ldsA[2][BM * LDT];
    __shared__ _Float16 ldsB[2][BN * LDT];

    const int tid  = threadIdx.x;
    const int lane = tid & 31;
    const int wv   = tid >> 5;            // 0..7
    const int wm   = (wv >> 2) << 6;      // 0 or 64  (M offset of wave)
    const int wn   = (wv & 3)  << 5;      // 0,32,64,96 (N offset of wave)

    const int m0 = blockIdx.y * BM;
    const int n0 = blockIdx.x * BN;

    // A-tile fill mapping: each thread moves 64 halfs of one half-row
    const int arow = tid >> 1;            // 0..127
    const int akh  = (tid & 1) << 6;      // 0 or 64
    const size_t rowoff = (size_t)(m0 + arow) * KDIM + akh;

    // B-tile fill mapping: 2 threads per column; each handles 4 int64 words
    const int bcol = tid & 127;
    const int bw0  = (tid >> 7) << 2;     // 0 or 4
    const int ng   = n0 + bcol;

    v8f acc[4][2];
    #pragma unroll
    for (int i = 0; i < 4; ++i)
        #pragma unroll
        for (int j = 0; j < 2; ++j)
            acc[i][j] = {};

    uint4    asta[8];
    uint64_t bsta[4];
    float    sN;
    int      zN;

    // prologue: stage + store tile 0
    load_stage_a<XHALF>(asta, xin, rowoff, 0);
    load_stage_b(bsta, sN, zN, qweight, scales, qzeros, 0, bw0, ng);
    store_stage_a(ldsA[0], arow, akh, asta);
    store_stage_b(ldsB[0], bcol, bw0, bsta, sN, zN);

    for (int t = 0; t < NT; ++t) {
        __syncthreads();                       // buf[t&1] ready for everyone
        if (t + 1 < NT) {                      // global loads overlap compute
            load_stage_a<XHALF>(asta, xin, rowoff, (t + 1) * BK);
            load_stage_b(bsta, sN, zN, qweight, scales, qzeros, t + 1, bw0, ng);
        }
        compute_tile(ldsA[t & 1], ldsB[t & 1], wm, wn, lane, acc);
        if (t + 1 < NT) {                      // write the *other* buffer
            store_stage_a(ldsA[(t + 1) & 1], arow, akh, asta);
            store_stage_b(ldsB[(t + 1) & 1], bcol, bw0, bsta, sN, zN);
        }
    }

    // ---- epilogue: bias add + store ----
    // D layout: VGPR v, lanes 0..15 -> M = v, N = lane; lanes 16..31 -> M = v+8
    #pragma unroll
    for (int mi = 0; mi < 4; ++mi) {
        #pragma unroll
        for (int ni = 0; ni < 2; ++ni) {
            const int ncol  = n0 + wn + (ni << 4) + (lane & 15);
            const float bv  = bias[ncol];
            const int mbase = m0 + wm + (mi << 4) + ((lane >> 4) << 3);
            #pragma unroll
            for (int v = 0; v < 8; ++v) {
                out[(size_t)(mbase + v) * NDIM + ncol] = acc[mi][ni][v] + bv;
            }
        }
    }
}

extern "C" void kernel_launch(void* const* d_in, const int* in_sizes, int n_in,
                              void* d_out, int out_size, void* d_ws, size_t ws_size,
                              hipStream_t stream) {
    const float*    x       = (const float*)d_in[0];
    const uint64_t* qweight = (const uint64_t*)d_in[1];
    const float*    scales  = (const float*)d_in[2];
    const uint64_t* qzeros  = (const uint64_t*)d_in[3];
    // d_in[4] = g_idx: by construction g_idx[k] == k/128; folded into indexing.
    const float*    bias    = (const float*)d_in[5];
    float*          out     = (float*)d_out;

    dim3 grid(NDIM / BN, TOKENS / BM);   // (32, 64)
    dim3 block(256);

    const size_t xh_bytes = (size_t)TOKENS * KDIM * sizeof(_Float16);  // 64 MB
    if (ws_size >= xh_bytes) {
        _Float16* xh = (_Float16*)d_ws;
        x_to_f16_kernel<<<(TOKENS * KDIM / 8) / 256, 256, 0, stream>>>(x, xh);
        gptq_wmma_kernel<true><<<grid, block, 0, stream>>>(
            xh, qweight, scales, qzeros, bias, out);
    } else {
        gptq_wmma_kernel<false><<<grid, block, 0, stream>>>(
            x, qweight, scales, qzeros, bias, out);
    }
}
